// DSAAttention_21199958573397
// MI455X (gfx1250) — matrix operand order repaired and checked by reading, not compiled, not run on hardware
//
#include <hip/hip_runtime.h>
#include <cstdint>

// ---------------------------------------------------------------------------
// DeepSeek-style sparse MLA attention for MI455X (gfx1250, wave32, WMMA).
// Matmuls on v_wmma_f32_16x16x32_bf16; fragments are contiguous b128 reads;
// non-transposed LDS tile fills use the CDNA5 async global->LDS engine
// (ASYNCcnt) when the toolchain exposes it.
// ---------------------------------------------------------------------------

#define S_LEN 2048
#define HID   2048
#define NH    16
#define DN    128
#define DR    64
#define DV    128
#define QLR   1024
#define KVLR  512
#define IH    8
#define IHD   64
#define TOPK  256
#define NEGF  -1000000000.0f

typedef unsigned short u16;
typedef __bf16  bf16_t;
typedef bf16_t  v16bf  __attribute__((ext_vector_type(16)));
typedef u16     u16x16 __attribute__((ext_vector_type(16)));
typedef float   v8f    __attribute__((ext_vector_type(8)));

union V16 {           // one WMMA 16xbf16 operand = two 16B chunks
  u16x16 h;
  v16bf  b;
  uint4  q[2];
  u16    e[16];
};

#if __has_builtin(__builtin_amdgcn_global_load_async_to_lds_b128) && \
    __has_builtin(__builtin_amdgcn_s_wait_asynccnt)
#define HAS_ASYNC_LDS 1
#else
#define HAS_ASYNC_LDS 0
#endif

// Builtin expects: (int __vector(4) AS1*, ... ) per hipcc diagnostic.
typedef int v4i_ __attribute__((vector_size(4 * sizeof(int))));
typedef __attribute__((address_space(1))) v4i_ as1_v4i;
typedef __attribute__((address_space(3))) v4i_ as3_v4i;

// 16B global -> LDS copy; async (ASYNCcnt) when available, else through VGPRs.
__device__ __forceinline__ void copy_b128_to_lds(const void* g, void* l) {
#if HAS_ASYNC_LDS
  __builtin_amdgcn_global_load_async_to_lds_b128(
      (as1_v4i*)(v4i_*)const_cast<void*>(g),
      (as3_v4i*)(v4i_*)l, 0, 0);
#else
  *(uint4*)l = *(const uint4*)g;
#endif
}
__device__ __forceinline__ void async_lds_wait() {
#if HAS_ASYNC_LDS
  __builtin_amdgcn_s_wait_asynccnt(0);
#endif
}

__device__ __forceinline__ u16 f2bf(float f) {
  unsigned u = __float_as_uint(f);
  unsigned r = u + 0x7FFFu + ((u >> 16) & 1u);
  return (u16)(r >> 16);
}
__device__ __forceinline__ v8f wmma_bf16(v16bf a, v16bf b, v8f c) {
  return __builtin_amdgcn_wmma_f32_16x16x32_bf16(false, a, false, b,
                                                 (short)0, c, false, false);
}
__device__ __forceinline__ v8f v8f_zero() {
  v8f z;
#pragma unroll
  for (int e = 0; e < 8; ++e) z[e] = 0.f;
  return z;
}

// ---------------------------------------------------------------------------
// fp32 -> bf16 bulk convert
// ---------------------------------------------------------------------------
__global__ void cvt_bf16_kernel(const float* __restrict__ in,
                                u16* __restrict__ out, size_t n) {
  size_t i = (size_t)blockIdx.x * blockDim.x + threadIdx.x;
  if (i < n) out[i] = f2bf(in[i]);
}

// ---------------------------------------------------------------------------
// Generic bf16 GEMM: C[M,N] = A[M,K] @ B[K,N], f32 accumulate.
// Requires M % 128 == 0 and K % 32 == 0 (true for every call below).
// N may be ragged (guarded). 256 thr = 8 waves; 128x128 tile; K step 32.
// A tile: async global->LDS b128. B tile: load-all-then-scatter transpose.
// ---------------------------------------------------------------------------
__global__ __launch_bounds__(256) void gemm_bf16_kernel(
    const u16* __restrict__ A, const u16* __restrict__ B,
    float* __restrict__ C, int M, int N, int K) {
  __shared__ u16 As[128][40];   // [row][k]   row stride 80B (16B aligned)
  __shared__ u16 Bt[128][40];   // [col][k]   transposed B tile

  const int tid  = threadIdx.x;
  const int lane = tid & 31;
  const int wid  = tid >> 5;
  const int wm   = wid >> 1;   // 0..3
  const int wn   = wid & 1;    // 0..1
  const int half = lane >> 4;
  const int l15  = lane & 15;
  const int bm   = blockIdx.x * 128;
  const int bn   = blockIdx.y * 128;
  const bool fullN = (bn + 128 <= N);

  v8f acc[2][4];
#pragma unroll
  for (int i = 0; i < 2; ++i)
#pragma unroll
    for (int j = 0; j < 4; ++j) acc[i][j] = v8f_zero();

  for (int k0 = 0; k0 < K; k0 += 32) {
    if (k0 + 32 < K) {  // warm GL2 for next tiles (global_prefetch_b8)
      __builtin_prefetch(&A[(size_t)(bm + (tid >> 1)) * K + k0 + 32], 0, 1);
      __builtin_prefetch(&B[(size_t)(k0 + 32 + (tid >> 3)) * N + bn], 0, 1);
    }
    // A tile: 128x32, two async b128 copies per thread (no VGPR staging)
#pragma unroll
    for (int i = 0; i < 2; ++i) {
      int idx = (tid + i * 256) * 8;
      int r = idx >> 5, c = idx & 31;
      copy_b128_to_lds(&A[(size_t)(bm + r) * K + k0 + c], &As[r][c]);
    }
    // B tile: 32x128 -> transposed into Bt[col][k].
    if (fullN) {
      uint4 bq[2];
#pragma unroll
      for (int i = 0; i < 2; ++i) {          // issue both loads first (MLP)
        int idx = (tid + i * 256) * 8;
        int r = idx >> 7, c = idx & 127;
        bq[i] = *(const uint4*)&B[(size_t)(k0 + r) * N + bn + c];
      }
#pragma unroll
      for (int i = 0; i < 2; ++i) {          // then scatter-transpose to LDS
        int idx = (tid + i * 256) * 8;
        int r = idx >> 7, c = idx & 127;
        union { uint4 q; u16 e[8]; } u;
        u.q = bq[i];
#pragma unroll
        for (int e = 0; e < 8; ++e) Bt[c + e][r] = u.e[e];
      }
    } else {
      for (int i = tid; i < 32 * 128; i += 256) {
        int r = i >> 7, c = i & 127;
        int gc = bn + c;
        Bt[c][r] = (gc < N) ? B[(size_t)(k0 + r) * N + gc] : (u16)0;
      }
    }
    async_lds_wait();
    __syncthreads();

    // A fragments: row fixed per lane; k chunks [kb,kb+8) u [kb+16,kb+24)
    v16bf afr[2];
#pragma unroll
    for (int i = 0; i < 2; ++i) {
      int row = wm * 32 + i * 16 + l15;
      int kb = half * 8;
      V16 f;
      f.q[0] = *(const uint4*)&As[row][kb];
      f.q[1] = *(const uint4*)&As[row][kb + 16];
      afr[i] = f.b;
    }
    // B fragments: col fixed per lane; k chunks [half*8,+8) u [half*8+16,+8)
    v16bf bfr[4];
#pragma unroll
    for (int j = 0; j < 4; ++j) {
      int col = wn * 64 + j * 16 + l15;
      V16 f;
      f.q[0] = *(const uint4*)&Bt[col][half * 8];
      f.q[1] = *(const uint4*)&Bt[col][half * 8 + 16];
      bfr[j] = f.b;
    }
#pragma unroll
    for (int i = 0; i < 2; ++i)
#pragma unroll
      for (int j = 0; j < 4; ++j)
        acc[i][j] = wmma_bf16(afr[i], bfr[j], acc[i][j]);
    __syncthreads();
  }

#pragma unroll
  for (int i = 0; i < 2; ++i)
#pragma unroll
    for (int j = 0; j < 4; ++j) {
      int col = bn + wn * 64 + j * 16 + l15;
      if (col >= N) continue;
#pragma unroll
      for (int v = 0; v < 8; ++v) {
        int row = bm + wm * 32 + i * 16 + v + half * 8;
        C[(size_t)row * N + col] = acc[i][j][v];
      }
    }
}

// ---------------------------------------------------------------------------
// Row-wise RMSNorm, fp32 in (strided) -> bf16 out (strided)
// ---------------------------------------------------------------------------
__global__ void rmsnorm_bf16_kernel(const float* __restrict__ in,
                                    const float* __restrict__ w,
                                    u16* __restrict__ out,
                                    int strideIn, int N, int strideOut) {
  int row = blockIdx.x;
  const float* xr = in + (size_t)row * strideIn;
  __shared__ float red[8];
  float ss = 0.f;
  for (int i = threadIdx.x; i < N; i += blockDim.x) { float v = xr[i]; ss += v * v; }
  for (int m = 16; m >= 1; m >>= 1) ss += __shfl_xor(ss, m, 32);
  if ((threadIdx.x & 31) == 0) red[threadIdx.x >> 5] = ss;
  __syncthreads();
  if (threadIdx.x == 0) {
    float t = 0.f;
    for (int i = 0; i < (int)(blockDim.x >> 5); ++i) t += red[i];
    red[0] = rsqrtf(t / (float)N + 1e-6f);
  }
  __syncthreads();
  float r = red[0];
  u16* o = out + (size_t)row * strideOut;
  for (int i = threadIdx.x; i < N; i += blockDim.x) o[i] = f2bf(xr[i] * r * w[i]);
}

// ---------------------------------------------------------------------------
// Q: interleaved RoPE on last 64 dims of each head, write [H][S][192] bf16
// ---------------------------------------------------------------------------
__global__ void rope_q_kernel(const float* __restrict__ q_pre,
                              const float* __restrict__ cosb,
                              const float* __restrict__ sinb,
                              u16* __restrict__ qbf) {
  int s = blockIdx.x;
  const float* qp = q_pre + (size_t)s * NH * (DN + DR);
  const float* cs = cosb + (size_t)s * DR;
  const float* sn = sinb + (size_t)s * DR;
  for (int idx = threadIdx.x; idx < NH * 192; idx += blockDim.x) {
    int h = idx / 192, d = idx % 192;
    float v;
    if (d < DN) {
      v = qp[h * 192 + d];
    } else {
      int dd = d - DN, i = dd >> 1;
      float c = cs[i], s_ = sn[i];
      float x1 = qp[h * 192 + DN + 2 * i];
      float x2 = qp[h * 192 + DN + 2 * i + 1];
      v = (dd & 1) ? (x2 * c + x1 * s_) : (x1 * c - x2 * s_);
    }
    qbf[((size_t)h * S_LEN + s) * 192 + d] = f2bf(v);
  }
}

// ---------------------------------------------------------------------------
// K/V assembly: k = [kvb_nope(128) | rope(k_pe)(64)], v = kvb[...,128:]
// ---------------------------------------------------------------------------
__global__ void assemble_kv_kernel(const float* __restrict__ kvb,
                                   const float* __restrict__ kv,
                                   const float* __restrict__ cosb,
                                   const float* __restrict__ sinb,
                                   u16* __restrict__ kbf, u16* __restrict__ vbf) {
  int s = blockIdx.x;
  const float* kb_ = kvb + (size_t)s * NH * (DN + DV);
  const float* pe  = kv + (size_t)s * (KVLR + DR) + KVLR;
  const float* cs  = cosb + (size_t)s * DR;
  const float* sn  = sinb + (size_t)s * DR;
  __shared__ float kpe[DR];
  for (int d = threadIdx.x; d < DR; d += blockDim.x) {
    int i = d >> 1;
    float c = cs[i], s_ = sn[i];
    float x1 = pe[2 * i], x2 = pe[2 * i + 1];
    kpe[d] = (d & 1) ? (x2 * c + x1 * s_) : (x1 * c - x2 * s_);
  }
  __syncthreads();
  for (int idx = threadIdx.x; idx < NH * 192; idx += blockDim.x) {
    int h = idx / 192, d = idx % 192;
    float v = (d < DN) ? kb_[h * 256 + d] : kpe[d - DN];
    kbf[((size_t)h * S_LEN + s) * 192 + d] = f2bf(v);
  }
  for (int idx = threadIdx.x; idx < NH * DV; idx += blockDim.x) {
    int h = idx / DV, d = idx % DV;
    vbf[((size_t)h * S_LEN + s) * DV + d] = f2bf(kb_[h * 256 + DN + d]);
  }
}

// ---------------------------------------------------------------------------
// gate[s][h] = x[s] . idx_gate[:,h]   (8 waves per block, one head each)
// ---------------------------------------------------------------------------
__global__ void gate_kernel(const float* __restrict__ x,
                            const float* __restrict__ idx_gate,
                            float* __restrict__ gate) {
  int s = blockIdx.x;
  int w = threadIdx.x >> 5, lane = threadIdx.x & 31;
  const float* xr = x + (size_t)s * HID;
  float acc = 0.f;
  for (int i = lane; i < HID; i += 32) acc += xr[i] * idx_gate[(size_t)i * IH + w];
  for (int m = 16; m >= 1; m >>= 1) acc += __shfl_xor(acc, m, 32);
  if (lane == 0) gate[s * IH + w] = acc;
}

// ---------------------------------------------------------------------------
// qmix[s][d] = (IH^-.5 * IHD^-.5) * sum_h gate[s][h] * rope_ni(qi)[s][h][d]
// ---------------------------------------------------------------------------
__global__ void qmix_kernel(const float* __restrict__ qi_pre,
                            const float* __restrict__ gate,
                            const float* __restrict__ cosb,
                            const float* __restrict__ sinb,
                            u16* __restrict__ qmix) {
  int s = blockIdx.x;
  int d = threadIdx.x;  // 0..63
  const float* qp = qi_pre + (size_t)s * IH * IHD;
  float c = cosb[(size_t)s * DR + d], s_ = sinb[(size_t)s * DR + d];
  float acc = 0.f;
#pragma unroll
  for (int h = 0; h < IH; ++h) {
    float xv  = qp[h * IHD + d];
    float rot = (d < 32) ? -qp[h * IHD + d + 32] : qp[h * IHD + d - 32];
    acc += gate[s * IH + h] * (xv * c + rot * s_);
  }
  qmix[(size_t)s * IHD + d] = f2bf(acc * 0.125f * 0.35355339059327f);
}

// ---------------------------------------------------------------------------
// ki: layernorm + non-interleaved rope, write TRANSPOSED bf16 [64][S]
// ---------------------------------------------------------------------------
__global__ void ki_kernel(const float* __restrict__ ki_pre,
                          const float* __restrict__ lw, const float* __restrict__ lb,
                          const float* __restrict__ cosb, const float* __restrict__ sinb,
                          u16* __restrict__ kit) {
  int s = blockIdx.x;
  int d = threadIdx.x;  // 0..63
  __shared__ float buf[IHD];
  __shared__ float mv[2];
  float x = ki_pre[(size_t)s * IHD + d];
  buf[d] = x;
  __syncthreads();
  if (d == 0) {
    float m = 0.f;
    for (int i = 0; i < IHD; ++i) m += buf[i];
    m /= (float)IHD;
    float v = 0.f;
    for (int i = 0; i < IHD; ++i) { float t = buf[i] - m; v += t * t; }
    v /= (float)IHD;
    mv[0] = m; mv[1] = rsqrtf(v + 1e-5f);
  }
  __syncthreads();
  float n = (x - mv[0]) * mv[1] * lw[d] + lb[d];
  __syncthreads();
  buf[d] = n;
  __syncthreads();
  float c = cosb[(size_t)s * DR + d], s_ = sinb[(size_t)s * DR + d];
  float rot = (d < 32) ? -buf[d + 32] : buf[d - 32];
  kit[(size_t)d * S_LEN + s] = f2bf(n * c + rot * s_);
}

// ---------------------------------------------------------------------------
// Top-K(256) per row via radix-select over order-flipped float bits.
// Emits key bitmask [S][64] u32.
// ---------------------------------------------------------------------------
__global__ __launch_bounds__(256) void topk_kernel(const float* __restrict__ iscore,
                                                   unsigned* __restrict__ maskbits) {
  int q = blockIdx.x;
  __shared__ unsigned ord[S_LEN];
  __shared__ int cnt;
  const float* row = iscore + (size_t)q * S_LEN;
  for (int i = threadIdx.x; i < S_LEN; i += 256) {
    float v = (i <= q) ? row[i] : NEGF;   // causal fold-in
    unsigned u = __float_as_uint(v);
    u ^= (u >> 31) ? 0xFFFFFFFFu : 0x80000000u;
    ord[i] = u;
  }
  __syncthreads();
  unsigned prefix = 0;
  int quota = TOPK;
  for (int bit = 31; bit >= 0; --bit) {
    unsigned bmask  = 1u << bit;
    unsigned himask = ~((bmask << 1) - 1u);
    if (threadIdx.x == 0) cnt = 0;
    __syncthreads();
    int local = 0;
    for (int i = threadIdx.x; i < S_LEN; i += 256) {
      unsigned u = ord[i];
      if ((u & himask) == prefix && (u & bmask)) local++;
    }
    atomicAdd(&cnt, local);
    __syncthreads();
    int c = cnt;
    if (c >= quota) prefix |= bmask; else quota -= c;
    __syncthreads();
  }
  for (int wdx = threadIdx.x; wdx < S_LEN / 32; wdx += 256) {
    unsigned bits = 0;
    for (int j = 0; j < 32; ++j)
      if (ord[wdx * 32 + j] >= prefix) bits |= (1u << j);
    maskbits[(size_t)q * (S_LEN / 32) + wdx] = bits;
  }
}

// ---------------------------------------------------------------------------
// Flash attention, one head x 64-query tile per WG. 4 waves x 16 rows.
// K tile: async global->LDS. V tile: load-all-then-scatter transpose.
// ---------------------------------------------------------------------------
__global__ __launch_bounds__(128) void flash_attn_kernel(
    const u16* __restrict__ Qb, const u16* __restrict__ Kb,
    const u16* __restrict__ Vb, const unsigned* __restrict__ maskbits,
    u16* __restrict__ Ob) {
  const int h    = blockIdx.y;
  const int qblk = blockIdx.x;
  const int tid  = threadIdx.x;
  const int lane = tid & 31, w = tid >> 5;
  const int half = lane >> 4, l15 = lane & 15;
  const int q0   = qblk * 64 + w * 16;

  const u16* Qh = Qb + (size_t)h * S_LEN * 192;
  const u16* Kh = Kb + (size_t)h * S_LEN * 192;
  const u16* Vh = Vb + (size_t)h * S_LEN * DV;

  __shared__ u16 Ks[32][200];    // [key][d], d-contiguous; row 400B
  __shared__ u16 Vst[128][40];   // [vcol][key], transposed
  __shared__ u16 Ps[4][16][32];  // per-wave P staging; row 64B

  // Q fragments (A layout): two contiguous 16B chunks per 32-d chunk
  v16bf qfr[6];
#pragma unroll
  for (int kc = 0; kc < 6; ++kc) {
    int m = q0 + l15;
    int kb = half * 8;
    V16 f;
    f.q[0] = *(const uint4*)&Qh[(size_t)m * 192 + kc * 32 + kb];
    f.q[1] = *(const uint4*)&Qh[(size_t)m * 192 + kc * 32 + kb + 16];
    qfr[kc] = f.b;
  }

  v8f Oacc[8];
#pragma unroll
  for (int nt = 0; nt < 8; ++nt) Oacc[nt] = v8f_zero();
  float mi[8], li[8];
#pragma unroll
  for (int v = 0; v < 8; ++v) { mi[v] = -3e38f; li[v] = 0.f; }

  const float scale = 0.07216878364870323f;  // (DN+DR)^-0.5
  const int nkb = (qblk + 1) * 2;            // 32-key blocks, causal bound

  for (int kb = 0; kb < nkb; ++kb) {
    const int kbase = kb * 32;
    __syncthreads();
    // K tile: 6 async b128 copies per thread, straight into LDS
#pragma unroll
    for (int i = 0; i < 6; ++i) {
      int idx = (tid + i * 128) * 8;
      int r = idx / 192, c = idx % 192;
      copy_b128_to_lds(&Kh[(size_t)(kbase + r) * 192 + c], &Ks[r][c]);
    }
    // V tile: load all 4 uint4 first, then scatter-transpose
    {
      uint4 vq[4];
#pragma unroll
      for (int i = 0; i < 4; ++i) {
        int idx = (tid + i * 128) * 8;
        int r = idx >> 7, c = idx & 127;
        vq[i] = *(const uint4*)&Vh[(size_t)(kbase + r) * DV + c];
      }
#pragma unroll
      for (int i = 0; i < 4; ++i) {
        int idx = (tid + i * 128) * 8;
        int r = idx >> 7, c = idx & 127;
        union { uint4 q; u16 e[8]; } u;
        u.q = vq[i];
#pragma unroll
        for (int e = 0; e < 8; ++e) Vst[c + e][r] = u.e[e];
      }
    }
    async_lds_wait();
    __syncthreads();

    // S tiles: 16 rows x 32 keys
    v8f Sv[2];
    Sv[0] = v8f_zero(); Sv[1] = v8f_zero();
#pragma unroll
    for (int t = 0; t < 2; ++t) {
      int key = t * 16 + l15;
#pragma unroll
      for (int kc = 0; kc < 6; ++kc) {
        V16 f;
        f.q[0] = *(const uint4*)&Ks[key][kc * 32 + half * 8];
        f.q[1] = *(const uint4*)&Ks[key][kc * 32 + half * 8 + 16];
        Sv[t] = wmma_bf16(qfr[kc], f.b, Sv[t]);
      }
    }

    // mask + online softmax
    float p[2][8];
#pragma unroll
    for (int v = 0; v < 8; ++v) {
      int row = q0 + v + half * 8;
      unsigned mb = maskbits[(size_t)row * (S_LEN / 32) + kb];
      float best = -3e38f;
#pragma unroll
      for (int t = 0; t < 2; ++t) {
        int key = kbase + t * 16 + l15;
        int bitpos = t * 16 + l15;
        float sv = Sv[t][v] * scale;
        if (key > row || !((mb >> bitpos) & 1u)) sv = NEGF;
        p[t][v] = sv;
        best = fmaxf(best, sv);
      }
      float rm = best;
      for (int m = 1; m < 16; m <<= 1) rm = fmaxf(rm, __shfl_xor(rm, m, 32));
      float mnew  = fmaxf(mi[v], rm);
      float alpha = __expf(mi[v] - mnew);
      mi[v] = mnew;
      float rsum = 0.f;
#pragma unroll
      for (int t = 0; t < 2; ++t) {
        float e = (p[t][v] < -0.5e9f) ? 0.f : __expf(p[t][v] - mnew);
        p[t][v] = e;
        rsum += e;
      }
      for (int m = 1; m < 16; m <<= 1) rsum += __shfl_xor(rsum, m, 32);
      li[v] = li[v] * alpha + rsum;
#pragma unroll
      for (int nt = 0; nt < 8; ++nt) Oacc[nt][v] *= alpha;
      Ps[w][v + half * 8][l15]      = f2bf(p[0][v]);
      Ps[w][v + half * 8][16 + l15] = f2bf(p[1][v]);
    }

    // P (A-frag, two 16B chunks) x V (B-frag, contiguous in Vst)
    V16 pf;
    pf.q[0] = *(const uint4*)&Ps[w][l15][half * 8];
    pf.q[1] = *(const uint4*)&Ps[w][l15][half * 8 + 16];
#pragma unroll
    for (int nt = 0; nt < 8; ++nt) {
      V16 vf;
      vf.q[0] = *(const uint4*)&Vst[nt * 16 + l15][half * 8];
      vf.q[1] = *(const uint4*)&Vst[nt * 16 + l15][half * 8 + 16];
      Oacc[nt] = wmma_bf16(pf.b, vf.b, Oacc[nt]);
    }
  }

  // epilogue: normalize, write [S][H*DV] bf16 for the output projection
#pragma unroll
  for (int v = 0; v < 8; ++v) {
    int row = q0 + v + half * 8;
    float inv = (li[v] > 0.f) ? (1.f / li[v]) : 0.f;
#pragma unroll
    for (int nt = 0; nt < 8; ++nt)
      Ob[(size_t)row * (NH * DV) + h * DV + nt * 16 + l15] = f2bf(Oacc[nt][v] * inv);
  }
}

// ---------------------------------------------------------------------------
// host-side launcher
// ---------------------------------------------------------------------------
static inline size_t alignup(size_t x) { return (x + 255) & ~(size_t)255; }

extern "C" void kernel_launch(void* const* d_in, const int* in_sizes, int n_in,
                              void* d_out, int out_size, void* d_ws, size_t ws_size,
                              hipStream_t stream) {
  (void)in_sizes; (void)n_in; (void)out_size; (void)ws_size;
  const float* x          = (const float*)d_in[0];
  const float* cosb       = (const float*)d_in[1];
  const float* sinb       = (const float*)d_in[2];
  /* attn_mask d_in[3] unused: causal mask applied analytically */
  const float* wq_a       = (const float*)d_in[4];
  const float* q_norm_w   = (const float*)d_in[5];
  const float* wq_b       = (const float*)d_in[6];
  const float* wkv_a      = (const float*)d_in[7];
  const float* kv_norm_w  = (const float*)d_in[8];
  const float* wkv_b      = (const float*)d_in[9];
  const float* wo         = (const float*)d_in[10];
  const float* idx_wq_b   = (const float*)d_in[11];
  const float* idx_wk     = (const float*)d_in[12];
  const float* idx_knorm_w= (const float*)d_in[13];
  const float* idx_knorm_b= (const float*)d_in[14];
  const float* idx_gate   = (const float*)d_in[15];
  float* out = (float*)d_out;

  char* p = (char*)d_ws;
  auto alloc = [&](size_t bytes) -> void* { void* r = p; p += alignup(bytes); return r; };

  u16* x_bf    = (u16*)alloc((size_t)S_LEN * HID * 2);
  u16* wqa_bf  = (u16*)alloc((size_t)HID * QLR * 2);
  u16* wqb_bf  = (u16*)alloc((size_t)QLR * NH * (DN + DR) * 2);
  u16* wkva_bf = (u16*)alloc((size_t)HID * (KVLR + DR) * 2);
  u16* wkvb_bf = (u16*)alloc((size_t)KVLR * NH * (DN + DV) * 2);
  u16* wo_bf   = (u16*)alloc((size_t)NH * DV * HID * 2);
  u16* iwqb_bf = (u16*)alloc((size_t)QLR * IH * IHD * 2);
  u16* iwk_bf  = (u16*)alloc((size_t)HID * IHD * 2);

  float* qr_f   = (float*)alloc((size_t)S_LEN * QLR * 4);
  u16*   qr_bf  = (u16*)  alloc((size_t)S_LEN * QLR * 2);
  float* q_f    = (float*)alloc((size_t)S_LEN * NH * (DN + DR) * 4);
  u16*   q_bf   = (u16*)  alloc((size_t)NH * S_LEN * 192 * 2);
  float* kv_f   = (float*)alloc((size_t)S_LEN * (KVLR + DR) * 4);
  u16*   kvn_bf = (u16*)  alloc((size_t)S_LEN * KVLR * 2);
  float* kvb_f  = (float*)alloc((size_t)S_LEN * NH * (DN + DV) * 4);
  u16*   k_bf   = (u16*)  alloc((size_t)NH * S_LEN * 192 * 2);
  u16*   v_bf   = (u16*)  alloc((size_t)NH * S_LEN * DV * 2);
  float* qi_f   = (float*)alloc((size_t)S_LEN * IH * IHD * 4);
  float* ki_f   = (float*)alloc((size_t)S_LEN * IHD * 4);
  float* gate_f = (float*)alloc((size_t)S_LEN * IH * 4);
  u16*   qmix_bf= (u16*)  alloc((size_t)S_LEN * IHD * 2);
  u16*   kit_bf = (u16*)  alloc((size_t)IHD * S_LEN * 2);
  float* isc_f  = (float*)alloc((size_t)S_LEN * S_LEN * 4);
  unsigned* mb  = (unsigned*)alloc((size_t)S_LEN * (S_LEN / 32) * 4);
  u16*   attn_bf= (u16*)  alloc((size_t)S_LEN * NH * DV * 2);

  auto cvt = [&](const float* in, u16* o, size_t n) {
    cvt_bf16_kernel<<<(unsigned)((n + 255) / 256), 256, 0, stream>>>(in, o, n);
  };
  auto gemm = [&](const u16* A, const u16* B, float* C, int M, int N, int K) {
    dim3 g((M + 127) / 128, (N + 127) / 128);
    gemm_bf16_kernel<<<g, 256, 0, stream>>>(A, B, C, M, N, K);
  };

  // convert activations + weights to bf16
  cvt(x, x_bf, (size_t)S_LEN * HID);
  cvt(wq_a, wqa_bf, (size_t)HID * QLR);
  cvt(wq_b, wqb_bf, (size_t)QLR * NH * (DN + DR));
  cvt(wkv_a, wkva_bf, (size_t)HID * (KVLR + DR));
  cvt(wkv_b, wkvb_bf, (size_t)KVLR * NH * (DN + DV));
  cvt(wo, wo_bf, (size_t)NH * DV * HID);
  cvt(idx_wq_b, iwqb_bf, (size_t)QLR * IH * IHD);
  cvt(idx_wk, iwk_bf, (size_t)HID * IHD);

  // q path
  gemm(x_bf, wqa_bf, qr_f, S_LEN, QLR, HID);
  rmsnorm_bf16_kernel<<<S_LEN, 256, 0, stream>>>(qr_f, q_norm_w, qr_bf, QLR, QLR, QLR);
  gemm(qr_bf, wqb_bf, q_f, S_LEN, NH * (DN + DR), QLR);
  rope_q_kernel<<<S_LEN, 256, 0, stream>>>(q_f, cosb, sinb, q_bf);

  // kv path
  gemm(x_bf, wkva_bf, kv_f, S_LEN, KVLR + DR, HID);
  rmsnorm_bf16_kernel<<<S_LEN, 256, 0, stream>>>(kv_f, kv_norm_w, kvn_bf, KVLR + DR, KVLR, KVLR);
  gemm(kvn_bf, wkvb_bf, kvb_f, S_LEN, NH * (DN + DV), KVLR);
  assemble_kv_kernel<<<S_LEN, 256, 0, stream>>>(kvb_f, kv_f, cosb, sinb, k_bf, v_bf);

  // indexer path (collapsed: index_score = (sum_h gate*qi) . ki^T)
  gemm(qr_bf, iwqb_bf, qi_f, S_LEN, IH * IHD, QLR);
  gemm(x_bf, iwk_bf, ki_f, S_LEN, IHD, HID);
  ki_kernel<<<S_LEN, IHD, 0, stream>>>(ki_f, idx_knorm_w, idx_knorm_b, cosb, sinb, kit_bf);
  gate_kernel<<<S_LEN, 256, 0, stream>>>(x, idx_gate, gate_f);
  qmix_kernel<<<S_LEN, IHD, 0, stream>>>(qi_f, gate_f, cosb, sinb, qmix_bf);
  gemm(qmix_bf, kit_bf, isc_f, S_LEN, S_LEN, IHD);
  topk_kernel<<<S_LEN, 256, 0, stream>>>(isc_f, mb);

  // flash attention
  flash_attn_kernel<<<dim3(S_LEN / 64, NH), 128, 0, stream>>>(q_bf, k_bf, v_bf, mb, attn_bf);

  // output projection
  gemm(attn_bf, wo_bf, out, S_LEN, HID, NH * DV);
}
